// HeterogeneousGraphTransformer_55980603736504
// MI455X (gfx1250) — matrix operand order repaired and checked by reading, not compiled, not run on hardware
//
#include <hip/hip_runtime.h>

// Problem sizes (compile-time constants from the reference)
#define BQ 8
#define SQ 2048
#define DQ 768
#define EQ 2

typedef __bf16 bf16;
typedef __attribute__((ext_vector_type(16))) bf16  v16bf;
typedef __attribute__((ext_vector_type(8)))  bf16  v8bf;
typedef __attribute__((ext_vector_type(8)))  float v8f;

#define TM 128
#define TN 128
#define TK 32
#define NTHREADS 256

// ---------------------------------------------------------------------------
// Optional CDNA5 async global->LDS staging (ASYNCcnt-tracked DMA), guarded so
// the file still compiles identically if this toolchain lacks the builtins.
// Builtin signature (from hipcc diagnostic): param0 is a pointer to
// 'int __attribute__((vector_size(16)))' in the global (AS1) address space.
// ---------------------------------------------------------------------------
#ifndef __has_builtin
#define __has_builtin(x) 0
#endif
#if __has_builtin(__builtin_amdgcn_global_load_async_to_lds_b128) && \
    __has_builtin(__builtin_amdgcn_s_wait_asynccnt)
#define USE_ASYNC_LDS 1
#define GLOBAL_AS __attribute__((address_space(1)))
#define LDS_AS    __attribute__((address_space(3)))
typedef int v4i __attribute__((vector_size(16)));   // GCC-vector, matches builtin
#else
#define USE_ASYNC_LDS 0
#endif

// ---------------------------------------------------------------------------
// LDS tile staging. A tile: [TM][TK] bf16 (rows of A, k contiguous).
// B tile stored TRANSPOSED: Bt[n][k] = B[k][n], so B-fragments are contiguous.
// ---------------------------------------------------------------------------
template <typename T>
__device__ __forceinline__ void stage(bf16 (*dst)[TK], const T* __restrict__ src,
                                      long ld, int tid) {
#pragma unroll
  for (int t = 0; t < (TM * TK) / NTHREADS; ++t) {
    int l = t * NTHREADS + tid;
    int r = l >> 5, c = l & 31;
    dst[r][c] = (bf16)src[(long)r * ld + c];
  }
}

template <typename T>
__device__ __forceinline__ void stage_t(bf16 (*dst)[TK], const T* __restrict__ src,
                                        long ld, int tid) {
  // dst[n][k] = src[k*ld + n]   (k: 0..31, n: 0..127); global reads coalesced in n
#pragma unroll
  for (int t = 0; t < (TN * TK) / NTHREADS; ++t) {
    int l = t * NTHREADS + tid;
    int k = l >> 7, n = l & 127;
    dst[n][k] = (bf16)src[(long)k * ld + n];
  }
}

// bf16 row-major tile copy: 128x32 bf16 = 512 x 16B chunks, 2 per thread.
// Async DMA variant when available (no VALU, overlaps with WMMA stream).
__device__ __forceinline__ void stage_bf16(bf16 (*dst)[TK], const bf16* __restrict__ src,
                                           long ld, int tid) {
#if USE_ASYNC_LDS
#pragma unroll
  for (int t = 0; t < (TM * TK) / (8 * NTHREADS); ++t) {  // 2 iterations
    int l = t * NTHREADS + tid;   // chunk id 0..511
    int r = l >> 2;               // row 0..127
    int c = (l & 3) << 3;         // col 0,8,16,24 (16B chunks)
    __builtin_amdgcn_global_load_async_to_lds_b128(
        (GLOBAL_AS v4i*)(src + (long)r * ld + c),
        (LDS_AS v4i*)&dst[r][c], 0, 0);
  }
#else
  stage(dst, src, ld, tid);
#endif
}

// Fence for tiles staged (possibly) via async DMA: drain own ASYNCcnt, then
// workgroup barrier so every wave's chunks are visible in LDS.
__device__ __forceinline__ void tile_fence() {
#if USE_ASYNC_LDS
  __builtin_amdgcn_s_wait_asynccnt(0);
#endif
  __syncthreads();
}

// ---------------------------------------------------------------------------
// WMMA fragment loads per ISA 7.12.2 (wave32).
// A 16x32 bf16: lane L holds row M=L%16; K = {kh..kh+7} U {16+kh..16+kh+7},
// kh = 8*(L/16)  -> two aligned 16B LDS loads.
// B 32x16 bf16: lane L holds col N=L%16; K = {K0..K0+7} U {16+K0..16+K0+7}, K0 = 8*(L/16)
// -> from the transposed tile these are two aligned 16B LDS loads.
// ---------------------------------------------------------------------------
__device__ __forceinline__ v16bf frag_a(const bf16 (*t)[TK], int row0, int lane) {
  int m  = row0 + (lane & 15);
  int kh = (lane >> 4) << 3;
  v8bf lo = *(const v8bf*)&t[m][kh];
  v8bf hi = *(const v8bf*)&t[m][16 + kh];
  v16bf r;
#pragma unroll
  for (int i = 0; i < 8; ++i) { r[i] = lo[i]; r[i + 8] = hi[i]; }
  return r;
}

__device__ __forceinline__ v16bf frag_b(const bf16 (*t)[TK], int col0, int lane) {
  int n  = col0 + (lane & 15);
  int k0 = (lane >> 4) << 3;
  v8bf lo = *(const v8bf*)&t[n][k0];
  v8bf hi = *(const v8bf*)&t[n][k0 + 16];
  v16bf r;
#pragma unroll
  for (int i = 0; i < 8; ++i) { r[i] = lo[i]; r[i + 8] = hi[i]; }
  return r;
}

// Each wave: 16 rows x 128 cols of the 128x128 tile -> 8 WMMAs per K-step.
__device__ __forceinline__ void mma_tile(const bf16 (*As)[TK], const bf16 (*Bs)[TK],
                                         v8f acc[8], int warp, int lane) {
  v16bf a = frag_a(As, warp * 16, lane);
#pragma unroll
  for (int j = 0; j < 8; ++j) {
    v16bf b = frag_b(Bs, j * 16, lane);
    acc[j] = __builtin_amdgcn_wmma_f32_16x16x32_bf16(false, a, false, b,
                                                     (short)0, acc[j], false, false);
  }
}

// ---------------------------------------------------------------------------
// K1: q/k/v[e] = x @ W[e]^T + b, output bf16.  grid = (M/128, D/128, 3*E)
// f32 sources need conversion -> VALU staging path.
// ---------------------------------------------------------------------------
__global__ __launch_bounds__(NTHREADS) void k_qkv(
    const float* __restrict__ x,
    const float* __restrict__ Wq, const float* __restrict__ bq,
    const float* __restrict__ Wk, const float* __restrict__ bk,
    const float* __restrict__ Wv, const float* __restrict__ bv,
    bf16* __restrict__ qo, bf16* __restrict__ ko, bf16* __restrict__ vo) {
  __shared__ __align__(16) bf16 As[TM][TK];
  __shared__ __align__(16) bf16 Bs[TN][TK];
  const int tid = threadIdx.x, lane = tid & 31, warp = tid >> 5;
  const int m0 = blockIdx.x * TM;      // over B*S
  const int n0 = blockIdx.y * TN;      // over D
  const int e = blockIdx.z & 1;
  const int which = blockIdx.z >> 1;   // 0=q 1=k 2=v
  const float* W = (which == 0 ? Wq : which == 1 ? Wk : Wv) + (size_t)e * DQ * DQ;
  const float* bias = (which == 0 ? bq : which == 1 ? bk : bv) + (size_t)e * DQ;
  bf16* out = (which == 0 ? qo : which == 1 ? ko : vo) + (size_t)e * BQ * SQ * DQ;

  v8f acc[8] = {};
  for (int kt = 0; kt < DQ / TK; ++kt) {
    const int k0 = kt * TK;
    __syncthreads();
    stage(As, x + (size_t)m0 * DQ + k0, DQ, tid);        // x rows -> bf16
    stage(Bs, W + (size_t)n0 * DQ + k0, DQ, tid);        // W[f,:] row == B^T row
    __syncthreads();
    if (kt + 1 < DQ / TK) {
      __builtin_prefetch(x + (size_t)(m0 + (tid >> 1)) * DQ + k0 + TK, 0, 1);
      __builtin_prefetch(W + (size_t)(n0 + (tid >> 1)) * DQ + k0 + TK, 0, 1);
    }
    mma_tile(As, Bs, acc, warp, lane);
  }
#pragma unroll
  for (int j = 0; j < 8; ++j) {
    const int n = n0 + j * 16 + (lane & 15);
    const float bb = bias[n];
#pragma unroll
    for (int i = 0; i < 8; ++i) {
      const int m = m0 + warp * 16 + ((lane >> 4) << 3) + i;
      out[(size_t)m * DQ + n] = (bf16)(acc[j][i] + bb);
    }
  }
}

// ---------------------------------------------------------------------------
// K2: scores = (Q K^T)/sqrt(D) * (adj+1e-6) + (adj==0 ? -1e4 : 0), f32 out.
// adj is read exactly once here (dominant HBM stream). Both tiles are bf16 in
// workspace -> async DMA staging. grid = (S/128, S/128, E*B)
// ---------------------------------------------------------------------------
__global__ __launch_bounds__(NTHREADS) void k_scores(
    const bf16* __restrict__ q, const bf16* __restrict__ k,
    const float* __restrict__ adj0, const float* __restrict__ adj1,
    float* __restrict__ scores) {
  __shared__ __align__(16) bf16 As[TM][TK];
  __shared__ __align__(16) bf16 Bs[TN][TK];
  const int tid = threadIdx.x, lane = tid & 31, warp = tid >> 5;
  const int t0 = blockIdx.x * TN;      // key block
  const int q0 = blockIdx.y * TM;      // query block
  const int e = blockIdx.z >> 3, b = blockIdx.z & 7;
  const size_t ebSD = ((size_t)e * BQ + b) * (size_t)SQ * DQ;
  const bf16* qb = q + ebSD;
  const bf16* kb = k + ebSD;
  const float* adj = (e ? adj1 : adj0) + (size_t)b * SQ * SQ;
  float* sc = scores + ((size_t)e * BQ + b) * (size_t)SQ * SQ;

  v8f acc[8] = {};
  for (int kt = 0; kt < DQ / TK; ++kt) {
    const int c0 = kt * TK;
    __syncthreads();
    stage_bf16(As, qb + (size_t)q0 * DQ + c0, DQ, tid);  // Q rows (async path)
    stage_bf16(Bs, kb + (size_t)t0 * DQ + c0, DQ, tid);  // K rows == (K^T)^T rows
    tile_fence();
    mma_tile(As, Bs, acc, warp, lane);
  }
  const float scale = rsqrtf((float)DQ);
#pragma unroll
  for (int j = 0; j < 8; ++j) {
    const int n = t0 + j * 16 + (lane & 15);
#pragma unroll
    for (int i = 0; i < 8; ++i) {
      const int m = q0 + warp * 16 + ((lane >> 4) << 3) + i;
      const float a = adj[(size_t)m * SQ + n];
      float s = acc[j][i] * scale;
      s = s * (a + 1e-6f) + (a == 0.0f ? -10000.0f : 0.0f);
      sc[(size_t)m * SQ + n] = s;
    }
  }
}

// ---------------------------------------------------------------------------
// K3: row softmax over 2048, probabilities written bf16. grid = E*B*S rows.
// ---------------------------------------------------------------------------
__global__ __launch_bounds__(NTHREADS) void k_softmax(const float* __restrict__ scores,
                                                      bf16* __restrict__ P) {
  const int row = blockIdx.x;
  const int tid = threadIdx.x;
  const float* src = scores + (size_t)row * SQ;
  bf16* dst = P + (size_t)row * SQ;
  __shared__ float red[NTHREADS];
  float vv[SQ / NTHREADS];
  float mx = -3.0e38f;
#pragma unroll
  for (int j = 0; j < SQ / NTHREADS; ++j) {
    vv[j] = src[tid + j * NTHREADS];
    mx = fmaxf(mx, vv[j]);
  }
  red[tid] = mx; __syncthreads();
  for (int s = NTHREADS / 2; s > 0; s >>= 1) {
    if (tid < s) red[tid] = fmaxf(red[tid], red[tid + s]);
    __syncthreads();
  }
  mx = red[0]; __syncthreads();
  float sum = 0.0f;
#pragma unroll
  for (int j = 0; j < SQ / NTHREADS; ++j) { vv[j] = __expf(vv[j] - mx); sum += vv[j]; }
  red[tid] = sum; __syncthreads();
  for (int s = NTHREADS / 2; s > 0; s >>= 1) {
    if (tid < s) red[tid] += red[tid + s];
    __syncthreads();
  }
  const float inv = 1.0f / red[0];
#pragma unroll
  for (int j = 0; j < SQ / NTHREADS; ++j)
    dst[tid + j * NTHREADS] = (bf16)(vv[j] * inv);
}

// ---------------------------------------------------------------------------
// K4: out = P @ V, written into concatenated layout cat[b,s,e*D+d] (bf16).
// A tile (P) is bf16 -> async path; V tile needs a transpose -> VALU staging.
// grid = (D/128, S/128, E*B).
// ---------------------------------------------------------------------------
__global__ __launch_bounds__(NTHREADS) void k_pv(const bf16* __restrict__ P,
                                                 const bf16* __restrict__ vmat,
                                                 bf16* __restrict__ cat) {
  __shared__ __align__(16) bf16 As[TM][TK];
  __shared__ __align__(16) bf16 Bs[TN][TK];
  const int tid = threadIdx.x, lane = tid & 31, warp = tid >> 5;
  const int n0 = blockIdx.x * TN;      // over D
  const int m0 = blockIdx.y * TM;      // over queries
  const int e = blockIdx.z >> 3, b = blockIdx.z & 7;
  const bf16* Pb = P + ((size_t)e * BQ + b) * (size_t)SQ * SQ;
  const bf16* Vb = vmat + ((size_t)e * BQ + b) * (size_t)SQ * DQ;

  v8f acc[8] = {};
  for (int kt = 0; kt < SQ / TK; ++kt) {
    const int k0 = kt * TK;
    __syncthreads();
    stage_bf16(As, Pb + (size_t)m0 * SQ + k0, SQ, tid);  // P rows (async path)
    stage_t(Bs, Vb + (size_t)k0 * DQ + n0, DQ, tid);     // Bt[n][k] = V[k][n]
    tile_fence();
    if (kt + 1 < SQ / TK)
      __builtin_prefetch(Pb + (size_t)(m0 + (tid >> 1)) * SQ + k0 + TK, 0, 1);
    mma_tile(As, Bs, acc, warp, lane);
  }
#pragma unroll
  for (int j = 0; j < 8; ++j) {
    const int n = n0 + j * 16 + (lane & 15);
#pragma unroll
    for (int i = 0; i < 8; ++i) {
      const int m = m0 + warp * 16 + ((lane >> 4) << 3) + i;
      cat[((size_t)b * SQ + m) * (EQ * DQ) + (size_t)e * DQ + n] = (bf16)acc[j][i];
    }
  }
}

// ---------------------------------------------------------------------------
// K5: y = cat @ Wo^T + bo + x (residual fused), f32 out. grid = (D/128, BS/128)
// cat tile is bf16 -> async path; Wo is f32 -> VALU staging.
// ---------------------------------------------------------------------------
__global__ __launch_bounds__(NTHREADS) void k_out(const bf16* __restrict__ cat,
                                                  const float* __restrict__ Wo,
                                                  const float* __restrict__ bo,
                                                  const float* __restrict__ x,
                                                  float* __restrict__ y) {
  __shared__ __align__(16) bf16 As[TM][TK];
  __shared__ __align__(16) bf16 Bs[TN][TK];
  const int tid = threadIdx.x, lane = tid & 31, warp = tid >> 5;
  const int n0 = blockIdx.x * TN;      // over D
  const int m0 = blockIdx.y * TM;      // over B*S
  const int KD = EQ * DQ;              // 1536

  v8f acc[8] = {};
  for (int kt = 0; kt < KD / TK; ++kt) {
    const int k0 = kt * TK;
    __syncthreads();
    stage_bf16(As, cat + (size_t)m0 * KD + k0, KD, tid); // cat rows (async path)
    stage(Bs, Wo + (size_t)n0 * KD + k0, KD, tid);       // Wo[f,:] row == B^T row
    tile_fence();
    mma_tile(As, Bs, acc, warp, lane);
  }
#pragma unroll
  for (int j = 0; j < 8; ++j) {
    const int n = n0 + j * 16 + (lane & 15);
    const float bb = bo[n];
#pragma unroll
    for (int i = 0; i < 8; ++i) {
      const int m = m0 + warp * 16 + ((lane >> 4) << 3) + i;
      y[(size_t)m * DQ + n] = acc[j][i] + bb + x[(size_t)m * DQ + n];
    }
  }
}

// ---------------------------------------------------------------------------
// K6: LayerNorm over D=768 per row. grid = B*S rows.
// ---------------------------------------------------------------------------
__global__ __launch_bounds__(NTHREADS) void k_ln(const float* __restrict__ y,
                                                 const float* __restrict__ gamma,
                                                 const float* __restrict__ beta,
                                                 float* __restrict__ out) {
  const int row = blockIdx.x, tid = threadIdx.x;
  const float* src = y + (size_t)row * DQ;
  float vv[DQ / NTHREADS];
  float s = 0.0f, ss = 0.0f;
#pragma unroll
  for (int j = 0; j < DQ / NTHREADS; ++j) {
    vv[j] = src[tid + j * NTHREADS];
    s += vv[j]; ss += vv[j] * vv[j];
  }
  __shared__ float r1[NTHREADS], r2[NTHREADS];
  r1[tid] = s; r2[tid] = ss; __syncthreads();
  for (int k = NTHREADS / 2; k > 0; k >>= 1) {
    if (tid < k) { r1[tid] += r1[tid + k]; r2[tid] += r2[tid + k]; }
    __syncthreads();
  }
  const float mean = r1[0] * (1.0f / DQ);
  const float var  = r2[0] * (1.0f / DQ) - mean * mean;
  const float rstd = rsqrtf(var + 1e-5f);
#pragma unroll
  for (int j = 0; j < DQ / NTHREADS; ++j) {
    const int c = tid + j * NTHREADS;
    out[(size_t)row * DQ + c] = (vv[j] - mean) * rstd * gamma[c] + beta[c];
  }
}

// ---------------------------------------------------------------------------
extern "C" void kernel_launch(void* const* d_in, const int* in_sizes, int n_in,
                              void* d_out, int out_size, void* d_ws, size_t ws_size,
                              hipStream_t stream) {
  (void)in_sizes; (void)n_in; (void)out_size; (void)ws_size;
  const float* x     = (const float*)d_in[0];
  const float* adj0  = (const float*)d_in[1];
  const float* adj1  = (const float*)d_in[2];
  const float* Wq    = (const float*)d_in[3];
  const float* bq    = (const float*)d_in[4];
  const float* Wk    = (const float*)d_in[5];
  const float* bk    = (const float*)d_in[6];
  const float* Wv    = (const float*)d_in[7];
  const float* bv    = (const float*)d_in[8];
  const float* Wo    = (const float*)d_in[9];
  const float* bo    = (const float*)d_in[10];
  const float* gamma = (const float*)d_in[11];
  const float* beta  = (const float*)d_in[12];
  float* out = (float*)d_out;

  // Workspace layout (bytes):
  //   [q|k|v bf16 : 3 * 50.3 MB][scores f32 : 268 MB][P bf16 : 134 MB]
  // cat (50.3 MB, bf16) and y (50.3 MB, f32) alias the scores region,
  // which is dead after the softmax pass. Total footprint ~554 MB.
  char* base = (char*)d_ws;
  const size_t EBSD = (size_t)EQ * BQ * SQ * DQ;
  const size_t EBSS = (size_t)EQ * BQ * SQ * SQ;
  bf16* qw = (bf16*)base;
  bf16* kw = qw + EBSD;
  bf16* vw = kw + EBSD;
  float* scores = (float*)(base + 3 * EBSD * sizeof(bf16));
  bf16*  P   = (bf16*)((char*)scores + EBSS * sizeof(float));
  bf16*  cat = (bf16*)scores;                          // alias (scores dead)
  float* y   = (float*)((char*)scores + ((size_t)64 << 20)); // after cat

  k_qkv    <<<dim3((BQ * SQ) / TM, DQ / TN, 3 * EQ), NTHREADS, 0, stream>>>(
      x, Wq, bq, Wk, bk, Wv, bv, qw, kw, vw);
  k_scores <<<dim3(SQ / TN, SQ / TM, EQ * BQ), NTHREADS, 0, stream>>>(
      qw, kw, adj0, adj1, scores);
  k_softmax<<<dim3(EQ * BQ * SQ), NTHREADS, 0, stream>>>(scores, P);
  k_pv     <<<dim3(DQ / TN, SQ / TM, EQ * BQ), NTHREADS, 0, stream>>>(P, vw, cat);
  k_out    <<<dim3(DQ / TN, (BQ * SQ) / TM), NTHREADS, 0, stream>>>(cat, Wo, bo, x, y);
  k_ln     <<<dim3(BQ * SQ), NTHREADS, 0, stream>>>(y, gamma, beta, out);
}
